// TwoTowerModel_26371099197499
// MI455X (gfx1250) — compile-verified, matrix operand order checked by reading, not run on hardware
//
#include <hip/hip_runtime.h>
#include <math.h>

// ---------------- model constants ----------------
constexpr int Bsz = 1024;   // batch
constexpr int Ssz = 128;    // history length
constexpr int Dm  = 256;    // model dim
constexpr int Dd  = 128;    // D (embedding dim)
constexpr int Hh  = 4;      // heads
constexpr int Dh  = 64;     // head dim
constexpr int Dff = 256;    // ffn dim
constexpr int Na  = 4;      // actions
constexpr int Cc  = 200;    // candidates
constexpr int BS  = Bsz * Ssz;   // 131072 rows
constexpr int BC  = Bsz * Cc;    // 204800 rows

typedef __attribute__((ext_vector_type(16))) _Float16 v16h;
typedef __attribute__((ext_vector_type(8)))  float    v8f;

// ---------------- block reductions ----------------
template<int NT>
__device__ inline float blk_sum(float v, float* sb) {
  int t = threadIdx.x;
  sb[t] = v; __syncthreads();
#pragma unroll
  for (int o = NT / 2; o > 0; o >>= 1) {
    if (t < o) sb[t] += sb[t + o];
    __syncthreads();
  }
  float r = sb[0]; __syncthreads();
  return r;
}
template<int NT>
__device__ inline float blk_max(float v, float* sb) {
  int t = threadIdx.x;
  sb[t] = v; __syncthreads();
#pragma unroll
  for (int o = NT / 2; o > 0; o >>= 1) {
    if (t < o) sb[t] = fmaxf(sb[t], sb[t + o]);
    __syncthreads();
  }
  float r = sb[0]; __syncthreads();
  return r;
}

__device__ inline float clean(float v) {  // nan_to_num(nan=0, posinf=1, neginf=-1)
  if (isnan(v)) return 0.f;
  if (isinf(v)) return v > 0.f ? 1.f : -1.f;
  return v;
}

// ---------------- WMMA fragment load (fp32 -> f16) ----------------
// Per-lane layout for 16x32 f16 A/B fragments (wave32):
//   lane<16 : elems 0..7 = K k0..k0+7,      elems 8..15 = K k0+16..k0+23
//   lane>=16: elems 0..7 = K k0+8..k0+15,   elems 8..15 = K k0+24..k0+31
__device__ inline v16h frag_load(const float* __restrict__ rowptr, int k0, int hi) {
  const float* p = rowptr + k0 + hi * 8;
  float4 f0 = *(const float4*)(p + 0);
  float4 f1 = *(const float4*)(p + 4);
  float4 f2 = *(const float4*)(p + 16);
  float4 f3 = *(const float4*)(p + 20);
  v16h a;
  a[0]=(_Float16)f0.x;  a[1]=(_Float16)f0.y;  a[2]=(_Float16)f0.z;  a[3]=(_Float16)f0.w;
  a[4]=(_Float16)f1.x;  a[5]=(_Float16)f1.y;  a[6]=(_Float16)f1.z;  a[7]=(_Float16)f1.w;
  a[8]=(_Float16)f2.x;  a[9]=(_Float16)f2.y;  a[10]=(_Float16)f2.z; a[11]=(_Float16)f2.w;
  a[12]=(_Float16)f3.x; a[13]=(_Float16)f3.y; a[14]=(_Float16)f3.z; a[15]=(_Float16)f3.w;
  return a;
}

// ---------------- fused WMMA GEMM ----------------
// out[m,n] = act( sum_k A[m,k]*W[n,k] + bias[n] + add1[m,n] + add2[m,n] )
// Each wave computes a 16x64 strip (1 M-tile x 4 N-tiles) to reuse A fragments 4x.
__global__ void __launch_bounds__(256)
wmma_gemm_kernel(const float* __restrict__ A, const float* __restrict__ W,
                 const float* __restrict__ bias,
                 const float* __restrict__ add1, const float* __restrict__ add2,
                 float* __restrict__ out, int M, int N, int K, int relu) {
  int wave = threadIdx.x >> 5;
  int lane = threadIdx.x & 31;
  int tilesN  = N >> 4;
  int groupsN = tilesN >> 2;                 // N-tiles processed 4 at a time
  int group   = blockIdx.x * 8 + wave;
  int totalG  = (M >> 4) * groupsN;
  if (group >= totalG) return;               // whole-wave uniform exit (EXEC stays all-1)
  int tm  = group / groupsN;
  int tn0 = (group % groupsN) * 4;
  int lr  = lane & 15;
  int hi  = lane >> 4;

  const float* arow = A + (size_t)(tm * 16 + lr) * K;
  const float* wrow0 = W + (size_t)((tn0 + 0) * 16 + lr) * K;
  const float* wrow1 = W + (size_t)((tn0 + 1) * 16 + lr) * K;
  const float* wrow2 = W + (size_t)((tn0 + 2) * 16 + lr) * K;
  const float* wrow3 = W + (size_t)((tn0 + 3) * 16 + lr) * K;

  v8f acc0 = {}, acc1 = {}, acc2 = {}, acc3 = {};
  for (int k0 = 0; k0 < K; k0 += 32) {
    v16h af = frag_load(arow, k0, hi);
    v16h b0 = frag_load(wrow0, k0, hi);
    acc0 = __builtin_amdgcn_wmma_f32_16x16x32_f16(false, af, false, b0, (short)0, acc0, false, false);
    v16h b1 = frag_load(wrow1, k0, hi);
    acc1 = __builtin_amdgcn_wmma_f32_16x16x32_f16(false, af, false, b1, (short)0, acc1, false, false);
    v16h b2 = frag_load(wrow2, k0, hi);
    acc2 = __builtin_amdgcn_wmma_f32_16x16x32_f16(false, af, false, b2, (short)0, acc2, false, false);
    v16h b3 = frag_load(wrow3, k0, hi);
    acc3 = __builtin_amdgcn_wmma_f32_16x16x32_f16(false, af, false, b3, (short)0, acc3, false, false);
  }

#pragma unroll
  for (int j = 0; j < 4; ++j) {
    v8f acc = (j == 0) ? acc0 : (j == 1) ? acc1 : (j == 2) ? acc2 : acc3;
    int n = (tn0 + j) * 16 + lr;
    float bv = bias ? bias[n] : 0.f;
#pragma unroll
    for (int i = 0; i < 8; ++i) {
      // C/D layout: VGPR i -> M = i (lanes 0-15) / i+8 (lanes 16-31); N = lane&15
      int m = tm * 16 + hi * 8 + i;
      size_t off = (size_t)m * N + n;
      float v = acc[i] + bv;
      if (add1) v += add1[off];
      if (add2) v += add2[off];
      if (relu) v = fmaxf(v, 0.f);
      out[off] = v;
    }
  }
}

// ---------------- padding mask ----------------
__global__ void pad_kernel(const int* __restrict__ hist_ids, int* __restrict__ pad) {
  __shared__ float red[Ssz];
  int b = blockIdx.x, s = threadIdx.x;
  int p = (hist_ids[b * Ssz + s] == 0) ? 1 : 0;
  float cnt = blk_sum<Ssz>((float)p, red);
  int all_pad = (cnt >= (float)Ssz - 0.5f);
  if (all_pad && s == 0) p = 0;  // unmask first position when all padded
  pad[b * Ssz + s] = p;
}

// ---------------- user embedding sum ----------------
__global__ void ue_kernel(const int* __restrict__ uid, const int* __restrict__ gen,
                          const int* __restrict__ age,
                          const float* __restrict__ user_emb, const float* __restrict__ gender_emb,
                          const float* __restrict__ age_emb, float* __restrict__ ue) {
  int b = blockIdx.x, d = threadIdx.x;
  ue[b * Dd + d] = user_emb[(size_t)uid[b] * Dd + d]
                 + gender_emb[(size_t)gen[b] * Dd + d]
                 + age_emb[(size_t)age[b] * Dd + d];
}

// ---------------- build sequence input x = concat(he, LN(actions@actW^T)) + pe --------
__global__ void build_x_kernel(const int* __restrict__ hist_ids, const float* __restrict__ actions,
                               const float* __restrict__ hist_emb, const float* __restrict__ act_W,
                               const float* __restrict__ act_b, const float* __restrict__ pe,
                               float* __restrict__ X) {
  __shared__ float red[Dd];
  int idx = blockIdx.x;             // b*S + s
  int s = idx % Ssz;
  int d = threadIdx.x;              // 0..127
  int id = hist_ids[idx];
  float he = hist_emb[(size_t)id * Dd + d];
  const float* act = actions + (size_t)idx * Na;
  float ae = act_b[d];
#pragma unroll
  for (int a = 0; a < Na; ++a) ae += act[a] * act_W[d * Na + a];
  float mean = blk_sum<Dd>(ae, red) * (1.f / Dd);
  float df = ae - mean;
  float var = blk_sum<Dd>(df * df, red) * (1.f / Dd);
  float ln = df * rsqrtf(var + 1e-6f);
  ln = fminf(fmaxf(ln, -10.f), 10.f);
  X[(size_t)idx * Dm + d]       = he + pe[s * Dm + d];
  X[(size_t)idx * Dm + Dd + d]  = ln + pe[s * Dm + Dd + d];
}

// ---------------- fused two-stage additive attention (one block per (b,h)) ------------
// Reads Q,K; writes u = gk * q into U (may alias K buffer: each block only touches its
// own (b,h) slice and finishes all K reads before the write loop barrier).
__global__ void fast_attn_kernel(const float* __restrict__ Q, const float* __restrict__ Km,
                                 const int* __restrict__ pad,
                                 const float* __restrict__ wa, const float* __restrict__ wb,
                                 float* __restrict__ U) {
  __shared__ float qs[Ssz * Dh];   // 32 KB
  __shared__ float red[Ssz];
  __shared__ float alpha[Ssz];
  __shared__ float gq[Dh];
  __shared__ float w2[Dh];
  __shared__ float gk[Dh];
  int bh = blockIdx.x;
  int b = bh / Hh, h = bh % Hh;
  int t = threadIdx.x;             // 128 threads
  const float scale = 0.125f;      // 1/sqrt(64)
  const float* qbase = Q  + (size_t)b * Ssz * Dm + h * Dh;
  const float* kbase = Km + (size_t)b * Ssz * Dm + h * Dh;

  for (int i = t; i < Ssz * Dh; i += Ssz) {
    int s = i / Dh, d = i % Dh;
    qs[i] = qbase[(size_t)s * Dm + d];
  }
  __syncthreads();

  int s = t;
  int masked = pad[b * Ssz + s];
  // stage 1: alpha = softmax(q . wa)
  float sa = 0.f;
  for (int d = 0; d < Dh; ++d) sa += qs[s * Dh + d] * wa[h * Dh + d];
  sa *= scale;
  if (masked) sa = -1e9f;
  float mx = blk_max<Ssz>(sa, red);
  float e  = expf(sa - mx);
  float se = blk_sum<Ssz>(e, red);
  alpha[s] = e / se;
  __syncthreads();
  if (t < Dh) {
    float g = 0.f;
    for (int ss = 0; ss < Ssz; ++ss) g += alpha[ss] * qs[ss * Dh + t];
    gq[t] = g;
    w2[t] = g * wb[h * Dh + t] * scale;   // fold wb*scale so sb = k . w2
  }
  __syncthreads();
  // stage 2: beta = softmax((gq*k) . wb)
  float sb2 = 0.f;
  const float* krow = kbase + (size_t)s * Dm;
  for (int d = 0; d < Dh; ++d) sb2 += krow[d] * w2[d];
  if (masked) sb2 = -1e9f;
  mx = blk_max<Ssz>(sb2, red);
  e  = expf(sb2 - mx);
  se = blk_sum<Ssz>(e, red);
  alpha[s] = e / se;                       // beta
  __syncthreads();
  if (t < Dh) {
    float g = 0.f;
    for (int ss = 0; ss < Ssz; ++ss) g += alpha[ss] * kbase[(size_t)ss * Dm + t];
    gk[t] = gq[t] * g;
  }
  __syncthreads();                         // all global K reads complete before writes
  float* ubase = U + (size_t)b * Ssz * Dm + h * Dh;
  for (int i = t; i < Ssz * Dh; i += Ssz) {
    int ss = i / Dh, d = i % Dh;
    ubase[(size_t)ss * Dm + d] = gk[d] * qs[i];
  }
}

// ---------------- layer norm over rows of Dm ----------------
__global__ void ln_kernel(const float* __restrict__ in, float* __restrict__ out,
                          const float* __restrict__ sc, const float* __restrict__ bi) {
  __shared__ float red[Dm];
  int row = blockIdx.x, t = threadIdx.x;
  float v = in[(size_t)row * Dm + t];
  float mean = blk_sum<Dm>(v, red) * (1.f / Dm);
  float d = v - mean;
  float var = blk_sum<Dm>(d * d, red) * (1.f / Dm);
  out[(size_t)row * Dm + t] = d * rsqrtf(var + 1e-5f) * sc[t] + bi[t];
}

// ---------------- masked mean pooling (with nan_to_num) ----------------
__global__ void pool_kernel(const float* __restrict__ X, const int* __restrict__ pad,
                            float* __restrict__ pooled) {
  int b = blockIdx.x, d = threadIdx.x;   // 256 threads
  float acc = 0.f, cnt = 0.f;
  for (int s = 0; s < Ssz; ++s) {
    float m = pad[b * Ssz + s] ? 0.f : 1.f;
    float v = clean(X[((size_t)b * Ssz + s) * Dm + d]);
    acc += v * m;
    cnt += m;
  }
  pooled[b * Dm + d] = acc / fmaxf(cnt, 1.f);
}

// ---------------- user repr normalize (l2n + second norm for un) ----------------
__global__ void user_norm_kernel(const float* __restrict__ urtmp, float* __restrict__ out_user,
                                 float* __restrict__ un) {
  __shared__ float red[Dd];
  int b = blockIdx.x, d = threadIdx.x;
  float v = clean(urtmp[b * Dd + d]);
  float n2 = blk_sum<Dd>(v * v, red);
  float urn = v / fmaxf(sqrtf(n2), 1e-6f);
  out_user[b * Dd + d] = urn;
  float n2b = blk_sum<Dd>(urn * urn, red);
  un[b * Dd + d] = urn / (sqrtf(n2b) + 1e-12f);
}

// ---------------- item embedding sum ----------------
__global__ void item_embed_kernel(const int* __restrict__ iids, const int* __restrict__ cats,
                                  const float* __restrict__ item_emb, const float* __restrict__ cat_emb,
                                  float* __restrict__ E) {
  int idx = blockIdx.x, d = threadIdx.x;
  E[(size_t)idx * Dd + d] = item_emb[(size_t)iids[idx] * Dd + d]
                          + cat_emb[(size_t)cats[idx] * Dd + d];
}

// ---------------- l2 normalize candidate rows ----------------
__global__ void l2n_kernel(const float* __restrict__ in, float* __restrict__ out) {
  __shared__ float red[Dd];
  int row = blockIdx.x, d = threadIdx.x;
  float v = in[(size_t)row * Dd + d];
  float n2 = blk_sum<Dd>(v * v, red);
  out[(size_t)row * Dd + d] = v / fmaxf(sqrtf(n2), 1e-6f);
}

// ---------------- final scoring: logits + combined-LN + unembed ----------------
__global__ void score_kernel(const float* __restrict__ cand, const float* __restrict__ un,
                             const float* __restrict__ onorm_s, const float* __restrict__ onorm_b,
                             const float* __restrict__ unemb_W, const float* __restrict__ unemb_b,
                             float* __restrict__ out_al, float* __restrict__ out_scores) {
  __shared__ float red[Dd];
  __shared__ float comb[Dd];
  int bc = blockIdx.x;
  int b = bc / Cc;
  int d = threadIdx.x;
  float cv = cand[(size_t)bc * Dd + d];
  float uv = un[b * Dd + d];
  float prod = uv * cv;
  float cn2 = blk_sum<Dd>(cv * cv, red);
  float inv = 1.f / (sqrtf(cn2) + 1e-12f);
  float dot = blk_sum<Dd>(prod, red) * inv;
  float score = fminf(fmaxf(dot, -100.f), 100.f);   // TEMPERATURE == 1
  float mean = blk_sum<Dd>(prod, red) * (1.f / Dd);
  float df = prod - mean;
  float var = blk_sum<Dd>(df * df, red) * (1.f / Dd);
  comb[d] = df * rsqrtf(var + 1e-5f) * onorm_s[d] + onorm_b[d];
  __syncthreads();
  if (d < Na) {
    float al = unemb_b[d];
    for (int j = 0; j < Dd; ++j) al += comb[j] * unemb_W[d * Dd + j];
    if (d == 0) al += score;
    al = fminf(fmaxf(al, -100.f), 100.f);
    out_al[(size_t)bc * Na + d] = al;
  }
  if (d == 0) out_scores[bc] = score;
}

// ======================= host launcher =======================
extern "C" void kernel_launch(void* const* d_in, const int* in_sizes, int n_in,
                              void* d_out, int out_size, void* d_ws, size_t ws_size,
                              hipStream_t stream) {
  (void)in_sizes; (void)n_in; (void)out_size; (void)ws_size;
  auto F = [&](int i) { return (const float*)d_in[i]; };
  auto I = [&](int i) { return (const int*)d_in[i]; };

  // setup_inputs insertion order for the 7 data inputs; then params flattened
  // as a JAX pytree (dict keys sorted alphabetically, list in order).
  const int* user_ids  = I(0);
  const int* hist_ids  = I(1);
  const float* actions = F(2);
  const int* gender    = I(3);
  const int* age       = I(4);
  const int* cand_ids  = I(5);
  const int* cand_cats = I(6);
  const float* act_W = F(7),  *act_b = F(8),  *age_emb = F(9),  *cat_emb = F(10);
  const float* gender_emb = F(11), *hist_emb = F(12), *item_emb = F(13);
  struct LayerP { const float *W1,*W2,*Wk,*Wq,*Wr,*b1,*b2,*bk,*bq,*br,*ln1b,*ln1s,*ln2b,*ln2s,*wa,*wb; };
  LayerP L[2];
  for (int l = 0; l < 2; ++l) {
    int base = 14 + 16 * l;
    L[l].W1 = F(base+0);  L[l].W2 = F(base+1);  L[l].Wk = F(base+2);  L[l].Wq = F(base+3);
    L[l].Wr = F(base+4);  L[l].b1 = F(base+5);  L[l].b2 = F(base+6);  L[l].bk = F(base+7);
    L[l].bq = F(base+8);  L[l].br = F(base+9);  L[l].ln1b = F(base+10); L[l].ln1s = F(base+11);
    L[l].ln2b = F(base+12); L[l].ln2s = F(base+13); L[l].wa = F(base+14); L[l].wb = F(base+15);
  }
  const float* mlp_W1 = F(46), *mlp_b1 = F(47), *mlp_W2 = F(48), *mlp_b2 = F(49);
  const float* onorm_b = F(50), *onorm_s = F(51), *out_W = F(52), *out_b = F(53);
  const float* pe = F(54), *unemb_W = F(55), *unemb_b = F(56), *user_emb = F(57);

  // workspace layout (floats)
  float* ws = (float*)d_ws;
  const size_t BIG = (size_t)BS * Dm;      // 33,554,432
  float* X  = ws;
  float* Qb = ws + BIG;
  float* Kb = ws + 2 * BIG;
  float* Ub = ws + 3 * BIG;
  float* sm = ws + 4 * BIG;
  int*   pad    = (int*)sm;                        // BS ints
  float* ue     = sm + BS;                         // B*Dd
  float* pooled = ue + (size_t)Bsz * Dd;           // B*Dm
  float* urtmp  = pooled + (size_t)Bsz * Dm;       // B*Dd
  float* un     = urtmp + (size_t)Bsz * Dd;        // B*Dd

  // d_out layout: al | scores | user_repr | cand_repr
  float* out = (float*)d_out;
  float* out_al     = out;
  float* out_scores = out + (size_t)Bsz * Cc * Na;
  float* out_user   = out_scores + (size_t)Bsz * Cc;
  float* out_cand   = out_user + (size_t)Bsz * Dd;

  auto gemm = [&](const float* A, const float* W, const float* bias,
                  const float* add1, const float* add2, float* o,
                  int M, int N, int K, int relu) {
    int groups = (M / 16) * (N / 64);       // 1x4 tile-groups per wave
    int blocks = (groups + 7) / 8;          // 8 waves / block
    wmma_gemm_kernel<<<blocks, 256, 0, stream>>>(A, W, bias, add1, add2, o, M, N, K, relu);
  };

  // ---- user tower ----
  pad_kernel<<<Bsz, Ssz, 0, stream>>>(hist_ids, pad);
  ue_kernel<<<Bsz, Dd, 0, stream>>>(user_ids, gender, age, user_emb, gender_emb, age_emb, ue);
  build_x_kernel<<<BS, Dd, 0, stream>>>(hist_ids, actions, hist_emb, act_W, act_b, pe, X);

  for (int l = 0; l < 2; ++l) {
    gemm(X, L[l].Wq, L[l].bq, nullptr, nullptr, Qb, BS, Dm, Dm, 0);            // q
    gemm(X, L[l].Wk, L[l].bk, nullptr, nullptr, Kb, BS, Dm, Dm, 0);            // k
    fast_attn_kernel<<<Bsz * Hh, Ssz, 0, stream>>>(Qb, Kb, pad, L[l].wa, L[l].wb, Kb); // u -> Kb
    gemm(Kb, L[l].Wr, L[l].br, Qb, X, Ub, BS, Dm, Dm, 0);                      // x+attn (+q)
    ln_kernel<<<BS, Dm, 0, stream>>>(Ub, X, L[l].ln1s, L[l].ln1b);
    gemm(X, L[l].W1, L[l].b1, nullptr, nullptr, Qb, BS, Dff, Dm, 1);           // relu ffn1
    gemm(Qb, L[l].W2, L[l].b2, X, nullptr, Kb, BS, Dm, Dff, 0);                // x+ffn
    ln_kernel<<<BS, Dm, 0, stream>>>(Kb, X, L[l].ln2s, L[l].ln2b);
  }

  pool_kernel<<<Bsz, Dm, 0, stream>>>(X, pad, pooled);
  gemm(pooled, out_W, out_b, ue, nullptr, urtmp, Bsz, Dd, Dm, 0);
  user_norm_kernel<<<Bsz, Dd, 0, stream>>>(urtmp, out_user, un);

  // ---- item tower (reuses the big buffers) ----
  item_embed_kernel<<<BC, Dd, 0, stream>>>(cand_ids, cand_cats, item_emb, cat_emb, X);
  gemm(X, mlp_W1, mlp_b1, nullptr, nullptr, Qb, BC, Dff, Dd, 1);   // spans Qb..Kb
  gemm(Qb, mlp_W2, mlp_b2, X, nullptr, Ub, BC, Dd, Dff, 0);
  l2n_kernel<<<BC, Dd, 0, stream>>>(Ub, out_cand);

  // ---- scoring ----
  score_kernel<<<BC, Dd, 0, stream>>>(out_cand, un, onorm_s, onorm_b,
                                      unemb_W, unemb_b, out_al, out_scores);
}
